// SimpleModel_56092272886017
// MI455X (gfx1250) — matrix operand order repaired and checked by reading, not compiled, not run on hardware
//
#include <hip/hip_runtime.h>
#include <hip/hip_bf16.h>
#include <math.h>

#define NUM_USER   100000
#define NUM_ITEM   100000
#define NUM_ASPECT 64
#define LATENT_DIM 64
#define BZ         1024
#define NUM_SAMPLE 1000

typedef __attribute__((ext_vector_type(16))) __bf16 v16bf;
typedef __attribute__((ext_vector_type(8)))  float  v8f;

union AFrag { v16bf v; __bf16 h[16]; };
union BFrag { v16bf v; uint4 q[2]; };

// Hardware RNE f32 -> bf16 (lowers to v_cvt_*bf16* on gfx1250, not the
// 3-op integer bit-twiddle sequence).
__device__ __forceinline__ __bf16 cvt_bf16(float f) { return (__bf16)f; }

// ---------------------------------------------------------------------------
// A-fragment (16x32 bf16, this wave supplies row m = lane%16).
// Lane layout: lanes<16 hold K = {0..7} u {16..23} (+k0); lanes>=16 shift +8.
// Both 8-element chunks are contiguous 32B -> two b128 loads each.
// ---------------------------------------------------------------------------
__device__ __forceinline__ void build_afrag(const float* __restrict__ row,
                                            const float* __restrict__ rel,
                                            int k0, int lane, AFrag& a) {
    int hi = (lane >> 4) * 8;
    const float4* r0 = (const float4*)(row + k0 + hi);
    const float4* r1 = (const float4*)(row + k0 + 16 + hi);
    const float4* e0 = (const float4*)(rel + k0 + hi);
    const float4* e1 = (const float4*)(rel + k0 + 16 + hi);
    float4 x0 = r0[0], x1 = r0[1], y0 = r1[0], y1 = r1[1];
    float4 u0 = e0[0], u1 = e0[1], v0 = e1[0], v1 = e1[1];
    a.h[0]  = cvt_bf16(x0.x + u0.x);
    a.h[1]  = cvt_bf16(x0.y + u0.y);
    a.h[2]  = cvt_bf16(x0.z + u0.z);
    a.h[3]  = cvt_bf16(x0.w + u0.w);
    a.h[4]  = cvt_bf16(x1.x + u1.x);
    a.h[5]  = cvt_bf16(x1.y + u1.y);
    a.h[6]  = cvt_bf16(x1.z + u1.z);
    a.h[7]  = cvt_bf16(x1.w + u1.w);
    a.h[8]  = cvt_bf16(y0.x + v0.x);
    a.h[9]  = cvt_bf16(y0.y + v0.y);
    a.h[10] = cvt_bf16(y0.z + v0.z);
    a.h[11] = cvt_bf16(y0.w + v0.w);
    a.h[12] = cvt_bf16(y1.x + v1.x);
    a.h[13] = cvt_bf16(y1.y + v1.y);
    a.h[14] = cvt_bf16(y1.z + v1.z);
    a.h[15] = cvt_bf16(y1.w + v1.w);
}

// ---------------------------------------------------------------------------
// B-fragment (32x16 bf16): lane L holds column n = L%16; lanes<16 K=k0+{0..7,16..23},
// lanes>=16 K=k0+{8..15,24..31} -> two 8-element chunks from row-major aspect row n.
// ---------------------------------------------------------------------------
__device__ __forceinline__ void load_bfrag(const __bf16* __restrict__ asp,
                                           int ntile, int k0, int lane, BFrag& b) {
    int n  = ntile * 16 + (lane & 15);
    int hi = (lane >> 4) * 8;
    const __bf16* p = asp + n * LATENT_DIM + k0 + hi;
    b.q[0] = *(const uint4*)(p);
    b.q[1] = *(const uint4*)(p + 16);
}

// ---------------------------------------------------------------------------
// aspect_emb f32 -> bf16 (8 KB table, reused by every GEMM tile)
// ---------------------------------------------------------------------------
__global__ void __launch_bounds__(256) cvt_aspect(const float* __restrict__ aspect,
                                                  __bf16* __restrict__ out) {
    int i = blockIdx.x * blockDim.x + threadIdx.x;
    if (i < NUM_ASPECT * LATENT_DIM) out[i] = cvt_bf16(aspect[i]);
}

// ---------------------------------------------------------------------------
// P = softmax((U+Ru) @ A^T)   one wave per 16 users; 8 WMMAs per wave.
// ---------------------------------------------------------------------------
__global__ void __launch_bounds__(32) compute_P(
        const int*   __restrict__ user_indices,
        const float* __restrict__ user_emb,
        const float* __restrict__ relation_emb,
        const __bf16* __restrict__ aspect_bf16,
        float* __restrict__ P) {
    int lane = threadIdx.x;
    int tile = blockIdx.x;
    int m15  = lane & 15;
    int uidx = user_indices[tile * 16 + m15];
    const float* row = user_emb + (long)uidx * LATENT_DIM;

    AFrag a0, a1;
    build_afrag(row, relation_emb /*Ru*/, 0,  lane, a0);
    build_afrag(row, relation_emb,        32, lane, a1);

    v8f acc[4];
#pragma unroll
    for (int t = 0; t < 4; ++t) {
        BFrag b0, b1;
        load_bfrag(aspect_bf16, t, 0,  lane, b0);
        load_bfrag(aspect_bf16, t, 32, lane, b1);
        v8f c = {};
        c = __builtin_amdgcn_wmma_f32_16x16x32_bf16(false, a0.v, false, b0.v, (short)0, c, false, false);
        c = __builtin_amdgcn_wmma_f32_16x16x32_bf16(false, a1.v, false, b1.v, (short)0, c, false, false);
        acc[t] = c;
    }

    int mrow = (lane >> 4) * 8;
#pragma unroll
    for (int r = 0; r < 8; ++r) {
        float x0 = acc[0][r], x1 = acc[1][r], x2 = acc[2][r], x3 = acc[3][r];
        float mx = fmaxf(fmaxf(x0, x1), fmaxf(x2, x3));
#pragma unroll
        for (int off = 1; off < 16; off <<= 1) mx = fmaxf(mx, __shfl_xor(mx, off, 32));
        float e0 = __expf(x0 - mx), e1 = __expf(x1 - mx);
        float e2 = __expf(x2 - mx), e3 = __expf(x3 - mx);
        float s = e0 + e1 + e2 + e3;
#pragma unroll
        for (int off = 1; off < 16; off <<= 1) s += __shfl_xor(s, off, 32);
        float inv = 1.0f / s;
        int user = tile * 16 + mrow + r;
        float* op = P + (long)user * NUM_ASPECT + m15;
        op[0]  = e0 * inv;
        op[16] = e1 * inv;
        op[32] = e2 * inv;
        op[48] = e3 * inv;
    }
}

// ---------------------------------------------------------------------------
// qdiff[item] = softmax((e+Rip)A^T) - softmax((e+Rin)A^T) for ALL 100K items.
// One wave per 16 items; 16 WMMAs per wave (2 relations x 4 N-tiles x 2 K).
// ---------------------------------------------------------------------------
__global__ void __launch_bounds__(32) compute_qdiff(
        const float* __restrict__ item_emb,
        const float* __restrict__ relation_emb,
        const __bf16* __restrict__ aspect_bf16,
        float* __restrict__ qdiff) {
    int lane = threadIdx.x;
    int tile = blockIdx.x;
    int m15  = lane & 15;
    const float* row = item_emb + ((long)tile * 16 + m15) * LATENT_DIM;

    BFrag b[4][2];
#pragma unroll
    for (int t = 0; t < 4; ++t) {
        load_bfrag(aspect_bf16, t, 0,  lane, b[t][0]);
        load_bfrag(aspect_bf16, t, 32, lane, b[t][1]);
    }

    float diff[4][8];
#pragma unroll
    for (int rel = 0; rel < 2; ++rel) {
        const float* R = relation_emb + (rel + 1) * LATENT_DIM;  // Rip, Rin
        AFrag a0, a1;
        build_afrag(row, R, 0,  lane, a0);
        build_afrag(row, R, 32, lane, a1);
        v8f acc[4];
#pragma unroll
        for (int t = 0; t < 4; ++t) {
            v8f c = {};
            c = __builtin_amdgcn_wmma_f32_16x16x32_bf16(false, a0.v, false, b[t][0].v, (short)0, c, false, false);
            c = __builtin_amdgcn_wmma_f32_16x16x32_bf16(false, a1.v, false, b[t][1].v, (short)0, c, false, false);
            acc[t] = c;
        }
#pragma unroll
        for (int r = 0; r < 8; ++r) {
            float x0 = acc[0][r], x1 = acc[1][r], x2 = acc[2][r], x3 = acc[3][r];
            float mx = fmaxf(fmaxf(x0, x1), fmaxf(x2, x3));
#pragma unroll
            for (int off = 1; off < 16; off <<= 1) mx = fmaxf(mx, __shfl_xor(mx, off, 32));
            float e0 = __expf(x0 - mx), e1 = __expf(x1 - mx);
            float e2 = __expf(x2 - mx), e3 = __expf(x3 - mx);
            float s = e0 + e1 + e2 + e3;
#pragma unroll
            for (int off = 1; off < 16; off <<= 1) s += __shfl_xor(s, off, 32);
            float inv = 1.0f / s;
            if (rel == 0) {
                diff[0][r] = e0 * inv; diff[1][r] = e1 * inv;
                diff[2][r] = e2 * inv; diff[3][r] = e3 * inv;
            } else {
                diff[0][r] -= e0 * inv; diff[1][r] -= e1 * inv;
                diff[2][r] -= e2 * inv; diff[3][r] -= e3 * inv;
            }
        }
    }

    int mrow = (lane >> 4) * 8;
#pragma unroll
    for (int r = 0; r < 8; ++r) {
        long item = (long)tile * 16 + mrow + r;
        float* op = qdiff + item * NUM_ASPECT + m15;
        op[0]  = diff[0][r];
        op[16] = diff[1][r];
        op[32] = diff[2][r];
        op[48] = diff[3][r];
    }
}

// ---------------------------------------------------------------------------
// ratings[b,n] = P[b] . qdiff[idx[b,n]] + bias, then softmax over n.
// Block per user; half-wave (16 lanes) per item: 256B coalesced row gather.
// ---------------------------------------------------------------------------
__global__ void __launch_bounds__(256) ratings_softmax(
        const int*   __restrict__ item_indices,
        const float* __restrict__ P,
        const float* __restrict__ qdiff,
        const float* __restrict__ item_bias,
        float* __restrict__ out) {
    __shared__ float Ps[NUM_ASPECT];
    __shared__ float rs[NUM_SAMPLE];
    __shared__ float red[256];
    int b = blockIdx.x, tid = threadIdx.x;
    if (tid < NUM_ASPECT) Ps[tid] = P[(long)b * NUM_ASPECT + tid];
    __syncthreads();

    int h = tid >> 4, l16 = tid & 15;
    float4 pv = *(const float4*)(Ps + l16 * 4);
    const int* idxrow = item_indices + (long)b * NUM_SAMPLE;
    for (int n = h; n < NUM_SAMPLE; n += 16) {
        int idx = idxrow[n];
        float4 qv = *(const float4*)(qdiff + (long)idx * NUM_ASPECT + l16 * 4);
        float part = pv.x * qv.x + pv.y * qv.y + pv.z * qv.z + pv.w * qv.w;
#pragma unroll
        for (int off = 1; off < 16; off <<= 1) part += __shfl_xor(part, off, 32);
        if (l16 == 0) rs[n] = part + item_bias[idx];
    }
    __syncthreads();

    // row max
    float mx = -INFINITY;
    for (int n = tid; n < NUM_SAMPLE; n += 256) mx = fmaxf(mx, rs[n]);
    red[tid] = mx; __syncthreads();
    for (int s = 128; s > 0; s >>= 1) {
        if (tid < s) red[tid] = fmaxf(red[tid], red[tid + s]);
        __syncthreads();
    }
    mx = red[0]; __syncthreads();

    // exponentiate in place (each n touched by exactly one thread), sum
    float sum = 0.0f;
    for (int n = tid; n < NUM_SAMPLE; n += 256) {
        float e = __expf(rs[n] - mx);
        rs[n] = e;
        sum += e;
    }
    red[tid] = sum; __syncthreads();
    for (int s = 128; s > 0; s >>= 1) {
        if (tid < s) red[tid] += red[tid + s];
        __syncthreads();
    }
    float inv = 1.0f / red[0];
    for (int n = tid; n < NUM_SAMPLE; n += 256)
        out[(long)b * NUM_SAMPLE + n] = rs[n] * inv;
}

// ---------------------------------------------------------------------------
// Fallback (only if workspace is too small): fused naive VALU version.
// ---------------------------------------------------------------------------
__global__ void __launch_bounds__(256) fused_naive(
        const int* __restrict__ user_indices, const int* __restrict__ item_indices,
        const float* __restrict__ user_emb,   const float* __restrict__ item_emb,
        const float* __restrict__ aspect_emb, const float* __restrict__ relation_emb,
        const float* __restrict__ item_bias,  float* __restrict__ out) {
    __shared__ float Us[LATENT_DIM];
    __shared__ float Ps[NUM_ASPECT];
    __shared__ float rs[NUM_SAMPLE];
    __shared__ float red[256];
    int b = blockIdx.x, tid = threadIdx.x;
    int uidx = user_indices[b];
    if (tid < LATENT_DIM) Us[tid] = user_emb[(long)uidx * LATENT_DIM + tid] + relation_emb[tid];
    __syncthreads();
    if (tid < NUM_ASPECT) {
        float d = 0;
        for (int k = 0; k < LATENT_DIM; ++k) d += Us[k] * aspect_emb[tid * LATENT_DIM + k];
        Ps[tid] = d;
    }
    __syncthreads();
    if (tid == 0) {
        float mx = Ps[0];
        for (int i = 1; i < NUM_ASPECT; ++i) mx = fmaxf(mx, Ps[i]);
        float s = 0;
        for (int i = 0; i < NUM_ASPECT; ++i) { Ps[i] = __expf(Ps[i] - mx); s += Ps[i]; }
        float inv = 1.0f / s;
        for (int i = 0; i < NUM_ASPECT; ++i) Ps[i] *= inv;
    }
    __syncthreads();
    for (int n = tid; n < NUM_SAMPLE; n += 256) {
        int idx = item_indices[(long)b * NUM_SAMPLE + n];
        const float* ir = item_emb + (long)idx * LATENT_DIM;
        float sump = 0, sumn = 0, accp = 0, accn = 0;
        for (int a = 0; a < NUM_ASPECT; ++a) {
            const float* ar = aspect_emb + a * LATENT_DIM;
            float dp = 0, dn = 0;
            for (int k = 0; k < LATENT_DIM; ++k) {
                float av = ar[k], iv = ir[k];
                dp += (iv + relation_emb[LATENT_DIM + k])     * av;
                dn += (iv + relation_emb[2 * LATENT_DIM + k]) * av;
            }
            float ep = __expf(dp), en = __expf(dn);
            sump += ep; sumn += en;
            accp += Ps[a] * ep; accn += Ps[a] * en;
        }
        rs[n] = accp / sump - accn / sumn + item_bias[idx];
    }
    __syncthreads();
    float mx = -INFINITY;
    for (int n = tid; n < NUM_SAMPLE; n += 256) mx = fmaxf(mx, rs[n]);
    red[tid] = mx; __syncthreads();
    for (int s = 128; s > 0; s >>= 1) { if (tid < s) red[tid] = fmaxf(red[tid], red[tid + s]); __syncthreads(); }
    mx = red[0]; __syncthreads();
    float sum = 0;
    for (int n = tid; n < NUM_SAMPLE; n += 256) { float e = __expf(rs[n] - mx); rs[n] = e; sum += e; }
    red[tid] = sum; __syncthreads();
    for (int s = 128; s > 0; s >>= 1) { if (tid < s) red[tid] += red[tid + s]; __syncthreads(); }
    float inv = 1.0f / red[0];
    for (int n = tid; n < NUM_SAMPLE; n += 256)
        out[(long)b * NUM_SAMPLE + n] = rs[n] * inv;
}

extern "C" void kernel_launch(void* const* d_in, const int* in_sizes, int n_in,
                              void* d_out, int out_size, void* d_ws, size_t ws_size,
                              hipStream_t stream) {
    const int*   user_indices = (const int*)  d_in[0];
    const int*   item_indices = (const int*)  d_in[1];
    const float* user_emb     = (const float*)d_in[2];
    const float* item_emb     = (const float*)d_in[3];
    const float* aspect_emb   = (const float*)d_in[4];
    const float* relation_emb = (const float*)d_in[5];
    const float* item_bias    = (const float*)d_in[6];
    float* out = (float*)d_out;

    const size_t off_P      = 8192;                                   // 64*64*2 rounded
    const size_t sz_P       = (size_t)BZ * NUM_ASPECT * sizeof(float);
    const size_t off_qdiff  = off_P + sz_P;
    const size_t need       = off_qdiff + (size_t)NUM_ITEM * NUM_ASPECT * sizeof(float);

    if (ws_size >= need) {
        __bf16* aspect_bf16 = (__bf16*)d_ws;
        float* P     = (float*)((char*)d_ws + off_P);
        float* qdiff = (float*)((char*)d_ws + off_qdiff);

        cvt_aspect     <<<16,            256, 0, stream>>>(aspect_emb, aspect_bf16);
        compute_P      <<<BZ / 16,        32, 0, stream>>>(user_indices, user_emb, relation_emb, aspect_bf16, P);
        compute_qdiff  <<<NUM_ITEM / 16,  32, 0, stream>>>(item_emb, relation_emb, aspect_bf16, qdiff);
        ratings_softmax<<<BZ,            256, 0, stream>>>(item_indices, P, qdiff, item_bias, out);
    } else {
        fused_naive<<<BZ, 256, 0, stream>>>(user_indices, item_indices, user_emb, item_emb,
                                            aspect_emb, relation_emb, item_bias, out);
    }
}
